// Attention_82291573392001
// MI455X (gfx1250) — compile-verified, matrix operand order checked by reading, not run on hardware
//
#include <hip/hip_runtime.h>

typedef __bf16 bf16_t;
typedef __attribute__((ext_vector_type(16))) __bf16 v16bf;
typedef __attribute__((ext_vector_type(8)))  __bf16 v8bf;
typedef __attribute__((ext_vector_type(8)))  float  v8f;

union FragBF {
    v16bf v;
    struct { v8bf lo; v8bf hi; } h;
};

#define DIMN     1024
#define HDS      16
#define HEAD_DIM 64
#define SEQ      2048
#define BATCH    4

// ---------------------------------------------------------------- convert
__global__ void cvt_f32_bf16(const float* __restrict__ in, bf16_t* __restrict__ out, int n) {
    int i = blockIdx.x * blockDim.x + threadIdx.x;
    int stride = gridDim.x * blockDim.x;
    for (; i < n; i += stride) out[i] = (bf16_t)in[i];
}

// ---------------------------------------------------------------- GEMM: C = A[M,K] * Bm[N,K]^T + bias
// block = 128 threads (4 waves), block tile 128x128, wave tile 64x64 (4x4 wmma)
// mode 0: QKV epilogue -> scatter q (scaled by 0.125), k into [bh,s,64]; v transposed [bh,64,s]
// mode 1: plain fp32 store to outf [M,N]
__global__ __launch_bounds__(128) void gemm_bf16_wmma(
    const bf16_t* __restrict__ A,
    const bf16_t* __restrict__ Bm,
    const float*  __restrict__ bias,
    int N, int mode,
    bf16_t* __restrict__ qout, bf16_t* __restrict__ kout, bf16_t* __restrict__ vtout,
    float* __restrict__ outf)
{
    const int K    = DIMN;
    const int lane = threadIdx.x & 31;
    const int wave = threadIdx.x >> 5;
    const int lr   = lane & 15;
    const int hl   = lane >> 4;
    const int c    = hl * 8;
    const int m0w  = blockIdx.x * 128 + (wave >> 1) * 64;
    const int n0w  = blockIdx.y * 128 + (wave & 1) * 64;

    v8f acc[4][4] = {};

    for (int kk = 0; kk < K; kk += 32) {
        FragBF af[4], bfm[4];
        #pragma unroll
        for (int mi = 0; mi < 4; ++mi) {
            const bf16_t* p = A + (size_t)(m0w + mi * 16 + lr) * K + kk + c;
            af[mi].h.lo = *(const v8bf*)p;
            af[mi].h.hi = *(const v8bf*)(p + 16);
        }
        #pragma unroll
        for (int ni = 0; ni < 4; ++ni) {
            const bf16_t* p = Bm + (size_t)(n0w + ni * 16 + lr) * K + kk + c;
            bfm[ni].h.lo = *(const v8bf*)p;
            bfm[ni].h.hi = *(const v8bf*)(p + 16);
        }
        #pragma unroll
        for (int mi = 0; mi < 4; ++mi)
            #pragma unroll
            for (int ni = 0; ni < 4; ++ni)
                acc[mi][ni] = __builtin_amdgcn_wmma_f32_16x16x32_bf16(
                    false, af[mi].v, false, bfm[ni].v, (short)0, acc[mi][ni], false, false);
    }

    #pragma unroll
    for (int mi = 0; mi < 4; ++mi) {
        #pragma unroll
        for (int ni = 0; ni < 4; ++ni) {
            #pragma unroll
            for (int r = 0; r < 8; ++r) {
                int row = m0w + mi * 16 + r + hl * 8;
                int col = n0w + ni * 16 + lr;
                float val = acc[mi][ni][r] + bias[col];
                if (mode == 0) {
                    int which = col >> 10;
                    int cc = col & 1023;
                    int h  = cc >> 6, hd = cc & 63;
                    int b  = row >> 11, s = row & 2047;
                    size_t bh = (size_t)b * HDS + h;
                    if (which == 0)
                        qout[(bh * SEQ + s) * HEAD_DIM + hd] = (bf16_t)(val * 0.125f);
                    else if (which == 1)
                        kout[(bh * SEQ + s) * HEAD_DIM + hd] = (bf16_t)val;
                    else
                        vtout[(bh * HEAD_DIM + hd) * SEQ + s] = (bf16_t)val;
                } else {
                    outf[(size_t)row * N + col] = val;
                }
            }
        }
    }
}

// ---------------------------------------------------------------- flash attention (causal)
// grid = (SEQ/64, HDS, BATCH), block = 128 (4 waves), wave = 16 query rows
__global__ __launch_bounds__(128) void flash_attn_wmma(
    const bf16_t* __restrict__ q,   // [B*H, S, 64]  (pre-scaled by 1/sqrt(hd))
    const bf16_t* __restrict__ k,   // [B*H, S, 64]
    const bf16_t* __restrict__ vt,  // [B*H, 64, S]
    bf16_t* __restrict__ o)         // [B, S, DIM]
{
    __shared__ bf16_t plds[4][16][32];

    const int lane = threadIdx.x & 31;
    const int wave = threadIdx.x >> 5;
    const int lr   = lane & 15;
    const int hl   = lane >> 4;
    const int c    = hl * 8;
    const int h    = blockIdx.y;
    const int b    = blockIdx.z;
    const size_t bh = (size_t)b * HDS + h;
    const int m0   = blockIdx.x * 64 + wave * 16;

    // Q fragments (d = 0..31 and 32..63)
    FragBF a0, a1;
    {
        const bf16_t* p = q + (bh * SEQ + (size_t)(m0 + lr)) * HEAD_DIM;
        a0.h.lo = *(const v8bf*)(p + c);
        a0.h.hi = *(const v8bf*)(p + c + 16);
        a1.h.lo = *(const v8bf*)(p + 32 + c);
        a1.h.hi = *(const v8bf*)(p + 32 + c + 16);
    }

    v8f oacc[4] = {};
    float mval[8], lsum[8];
    #pragma unroll
    for (int r = 0; r < 8; ++r) { mval[r] = -__builtin_inff(); lsum[r] = 0.0f; }

    for (int j = 0; j <= m0 + 15; j += 32) {
        v8f s0 = {}, s1 = {};
        {
            const bf16_t* p = k + (bh * SEQ + (size_t)(j + lr)) * HEAD_DIM;
            FragBF b0, b1;
            b0.h.lo = *(const v8bf*)(p + c);       b0.h.hi = *(const v8bf*)(p + c + 16);
            b1.h.lo = *(const v8bf*)(p + 32 + c);  b1.h.hi = *(const v8bf*)(p + 32 + c + 16);
            s0 = __builtin_amdgcn_wmma_f32_16x16x32_bf16(false, a0.v, false, b0.v, (short)0, s0, false, false);
            s0 = __builtin_amdgcn_wmma_f32_16x16x32_bf16(false, a1.v, false, b1.v, (short)0, s0, false, false);
        }
        {
            const bf16_t* p = k + (bh * SEQ + (size_t)(j + 16 + lr)) * HEAD_DIM;
            FragBF b0, b1;
            b0.h.lo = *(const v8bf*)(p + c);       b0.h.hi = *(const v8bf*)(p + c + 16);
            b1.h.lo = *(const v8bf*)(p + 32 + c);  b1.h.hi = *(const v8bf*)(p + 32 + c + 16);
            s1 = __builtin_amdgcn_wmma_f32_16x16x32_bf16(false, a0.v, false, b0.v, (short)0, s1, false, false);
            s1 = __builtin_amdgcn_wmma_f32_16x16x32_bf16(false, a1.v, false, b1.v, (short)0, s1, false, false);
        }

        // online softmax per row (row data spread across 16-lane halves)
        #pragma unroll
        for (int r = 0; r < 8; ++r) {
            int qrow = m0 + r + hl * 8;
            float v0 = s0[r]; if (j + lr > qrow)      v0 = -__builtin_inff();
            float v1 = s1[r]; if (j + 16 + lr > qrow) v1 = -__builtin_inff();
            float rm = fmaxf(v0, v1);
            rm = fmaxf(rm, __shfl_xor(rm, 1, 32));
            rm = fmaxf(rm, __shfl_xor(rm, 2, 32));
            rm = fmaxf(rm, __shfl_xor(rm, 4, 32));
            rm = fmaxf(rm, __shfl_xor(rm, 8, 32));
            float mnew = fmaxf(mval[r], rm);
            float corr = __expf(mval[r] - mnew);
            float e0 = __expf(v0 - mnew);
            float e1 = __expf(v1 - mnew);
            float rs = e0 + e1;
            rs += __shfl_xor(rs, 1, 32);
            rs += __shfl_xor(rs, 2, 32);
            rs += __shfl_xor(rs, 4, 32);
            rs += __shfl_xor(rs, 8, 32);
            lsum[r] = lsum[r] * corr + rs;
            mval[r] = mnew;
            #pragma unroll
            for (int dt = 0; dt < 4; ++dt) oacc[dt][r] *= corr;
            plds[wave][r + hl * 8][lr]      = (bf16_t)e0;
            plds[wave][r + hl * 8][16 + lr] = (bf16_t)e1;
        }

        // wave-private LDS transpose: C-layout P -> A-layout fragment (16x32 over keys)
        asm volatile("s_wait_dscnt 0" ::: "memory");
        FragBF pf;
        pf.h.lo = *(const v8bf*)&plds[wave][lr][c];
        pf.h.hi = *(const v8bf*)&plds[wave][lr][c + 16];

        #pragma unroll
        for (int dt = 0; dt < 4; ++dt) {
            const bf16_t* p = vt + (bh * HEAD_DIM + (size_t)(dt * 16 + lr)) * SEQ + j;
            FragBF vf;
            vf.h.lo = *(const v8bf*)(p + c);
            vf.h.hi = *(const v8bf*)(p + c + 16);
            oacc[dt] = __builtin_amdgcn_wmma_f32_16x16x32_bf16(
                false, pf.v, false, vf.v, (short)0, oacc[dt], false, false);
        }
    }

    #pragma unroll
    for (int r = 0; r < 8; ++r) {
        int qrow = m0 + r + hl * 8;
        float inv = 1.0f / lsum[r];
        #pragma unroll
        for (int dt = 0; dt < 4; ++dt)
            o[((size_t)b * SEQ + qrow) * DIMN + h * HEAD_DIM + dt * 16 + lr] =
                (bf16_t)(oacc[dt][r] * inv);
    }
}

// ---------------------------------------------------------------- launcher
extern "C" void kernel_launch(void* const* d_in, const int* in_sizes, int n_in,
                              void* d_out, int out_size, void* d_ws, size_t ws_size,
                              hipStream_t stream)
{
    const float* x     = (const float*)d_in[0];  // [4,2048,1024]
    const float* w_qkv = (const float*)d_in[1];  // [3072,1024]
    const float* b_qkv = (const float*)d_in[2];  // [3072]
    const float* w_out = (const float*)d_in[3];  // [1024,1024]
    const float* b_out = (const float*)d_in[4];  // [1024]
    float* out = (float*)d_out;                  // [4,2048,1024]

    const int M   = BATCH * SEQ;     // 8192
    const int nX  = M * DIMN;        // 8388608
    const int nWq = 3 * DIMN * DIMN; // 3145728
    const int nWo = DIMN * DIMN;     // 1048576
    const int nHd = BATCH * HDS * SEQ * HEAD_DIM; // 8388608

    char* ws = (char*)d_ws;
    size_t off = 0;
    bf16_t* x_bf  = (bf16_t*)(ws + off); off += (size_t)nX  * 2;
    bf16_t* wq_bf = (bf16_t*)(ws + off); off += (size_t)nWq * 2;
    bf16_t* wo_bf = (bf16_t*)(ws + off); off += (size_t)nWo * 2;
    bf16_t* q_bf  = (bf16_t*)(ws + off); off += (size_t)nHd * 2;
    bf16_t* k_bf  = (bf16_t*)(ws + off); off += (size_t)nHd * 2;
    bf16_t* vt_bf = (bf16_t*)(ws + off); off += (size_t)nHd * 2;
    bf16_t* o_bf  = (bf16_t*)(ws + off); off += (size_t)nX  * 2;

    cvt_f32_bf16<<<(nX  + 255) / 256, 256, 0, stream>>>(x,     x_bf,  nX);
    cvt_f32_bf16<<<(nWq + 255) / 256, 256, 0, stream>>>(w_qkv, wq_bf, nWq);
    cvt_f32_bf16<<<(nWo + 255) / 256, 256, 0, stream>>>(w_out, wo_bf, nWo);

    // QKV projection: [8192,1024] x [3072,1024]^T
    dim3 g1(M / 128, (3 * DIMN) / 128);
    gemm_bf16_wmma<<<g1, 128, 0, stream>>>(x_bf, wq_bf, b_qkv, 3 * DIMN, 0,
                                           q_bf, k_bf, vt_bf, nullptr);

    // causal flash attention
    dim3 g2(SEQ / 64, HDS, BATCH);
    flash_attn_wmma<<<g2, 128, 0, stream>>>(q_bf, k_bf, vt_bf, o_bf);

    // output projection: [8192,1024] x [1024,1024]^T -> fp32 d_out
    dim3 g3(M / 128, DIMN / 128);
    gemm_bf16_wmma<<<g3, 128, 0, stream>>>(o_bf, wo_bf, b_out, DIMN, 1,
                                           nullptr, nullptr, nullptr, out);
}
